// RNNAutoEncoderModel_56392920596775
// MI455X (gfx1250) — compile-verified, matrix operand order checked by reading, not run on hardware
//
#include <hip/hip_runtime.h>
#include <hip/hip_bf16.h>

// ---------------------------------------------------------------------------
// RNN autoencoder (2-layer LSTM enc/dec, tied embedding) for MI455X / gfx1250.
//   * hoist x@Wih^T out of the time loop as large bf16 WMMA GEMMs
//   * sequential part = 512 FUSED step kernels: [16,1024]x[1024,4096] WMMA
//     GEMM + LSTM pointwise epilogue entirely in registers (each wave owns
//     the i/f/g/o tiles for the same 16 hidden columns), with a rolled
//     ping-pong pipelined K loop so WMMAs cover the L2 load latency.
//     Whh is 8.4 MB bf16 -> L2-resident; h double-buffered across steps.
//   * tied projection [2048,1024]x[1024,32000] as one big bf16 WMMA GEMM
//   * fp32 accumulation everywhere (v_wmma_f32_16x16x32_bf16)
// ---------------------------------------------------------------------------

typedef __attribute__((ext_vector_type(16))) __bf16 bf16x16;
typedef __attribute__((ext_vector_type(8)))  __bf16 bf16x8;
typedef __attribute__((ext_vector_type(8)))  float  f32x8;

#define S_  128
#define B_  16
#define V_  32000
#define H_  1024
#define G4_ 4096   // 4*H
#define SB_ 2048   // S*B

__device__ __forceinline__ float sigmoidf_(float x) {
    return 1.0f / (1.0f + __expf(-x));
}

__device__ __forceinline__ bf16x16 load_frag16(const __bf16* p) {
    const bf16x8 lo = *(const bf16x8*)(p);
    const bf16x8 hi = *(const bf16x8*)(p + 16);
    return __builtin_shufflevector(lo, hi, 0, 1, 2, 3, 4, 5, 6, 7,
                                   8, 9, 10, 11, 12, 13, 14, 15);
}

__device__ __forceinline__ f32x8 wmma_bf16(const bf16x16 a, const bf16x16 b,
                                           const f32x8 cacc) {
    return __builtin_amdgcn_wmma_f32_16x16x32_bf16(
        false, a, false, b, (short)0, cacc, false, false);
}

// ---------------------------------------------------------------------------
// WMMA GEMM:  C[M,N](f32) = A[M,K](bf16) * W[N,K]^T(bf16) + bias(opt)
// Both operands are row-major with K contiguous, so A and W fragments load
// identically per the ISA 16-bit A layout: lanes 0-15 take K {0..7,16..23},
// lanes 16-31 take K {8..15,24..31} (two 16-byte loads per fragment).
// Each wave owns a (16*MSUB) x (16*NSUB) register tile; W fragments shared
// across MSUB row-tiles. Requirements (all call sites satisfy them):
// M % (16*MSUB)==0, N % (16*NSUB)==0 for every strip below N, K % 32 == 0.
// ---------------------------------------------------------------------------
template <int MSUB, int NSUB>
__global__ __launch_bounds__(256) void wmma_gemm_bf16(
    const __bf16* __restrict__ A, const __bf16* __restrict__ W,
    const float* __restrict__ bias, float* __restrict__ C,
    int M, int N, int K)
{
    const int lane = threadIdx.x & 31;
    // wave index is wave-uniform: force scalar so guards become s_cbranch
    // (EXEC stays all-1s around every WMMA).
    const int wave = __builtin_amdgcn_readfirstlane((int)threadIdx.x >> 5);
    const int m0   = blockIdx.y * (16 * MSUB);
    const int n0   = (blockIdx.x * 8 + wave) * (16 * NSUB);
    if (n0 >= N) return;   // scalar branch; only partial last block hits this

    const int half = lane >> 4;   // 0: lanes 0-15, 1: lanes 16-31
    const int l16  = lane & 15;

    const __bf16* aptr[MSUB];
#pragma unroll
    for (int mi = 0; mi < MSUB; ++mi)
        aptr[mi] = A + (size_t)(m0 + mi * 16 + l16) * K + half * 8;

    f32x8 acc[MSUB][NSUB];
#pragma unroll
    for (int mi = 0; mi < MSUB; ++mi)
#pragma unroll
        for (int t = 0; t < NSUB; ++t) {
            const float base = bias ? bias[n0 + t * 16 + l16] : 0.0f;
#pragma unroll
            for (int r = 0; r < 8; ++r) acc[mi][t][r] = base;
        }

    for (int k0 = 0; k0 < K; k0 += 32) {
        bf16x16 afrag[MSUB];
#pragma unroll
        for (int mi = 0; mi < MSUB; ++mi) {
            if (k0 + 64 < K) __builtin_prefetch(aptr[mi] + k0 + 64, 0, 3);
            afrag[mi] = load_frag16(aptr[mi] + k0);
        }
#pragma unroll
        for (int t = 0; t < NSUB; ++t) {
            const bf16x16 wfrag =
                load_frag16(W + (size_t)(n0 + t * 16 + l16) * K + k0 + half * 8);
#pragma unroll
            for (int mi = 0; mi < MSUB; ++mi)
                acc[mi][t] = wmma_bf16(afrag[mi], wfrag, acc[mi][t]);
        }
    }

#pragma unroll
    for (int mi = 0; mi < MSUB; ++mi) {
        const int crow = m0 + mi * 16 + half * 8;
#pragma unroll
        for (int t = 0; t < NSUB; ++t) {
            const int ccol = n0 + t * 16 + l16;
#pragma unroll
            for (int r = 0; r < 8; ++r)
                C[(size_t)(crow + r) * N + ccol] = acc[mi][t][r];
        }
    }
}

// ---------------------------------------------------------------------------
// Fused LSTM step:  g = hprev @ Whh^T + Gx[s]   (Gx = x@Wih^T + bih + bhh)
//                   c' = sig(f)*c + sig(i)*tanh(g);  h' = sig(o)*tanh(c')
// Each wave owns the four gate tiles {j0, H+j0, 2H+j0, 3H+j0} x 16 batch
// rows, so the whole cell update runs in registers after the WMMA chain.
// K loop is a ROLLED ping-pong pipeline (unroll disabled): each iteration
// loads fragment set B @ k+32, consumes set A @ k, loads set A @ k+64,
// consumes set B @ k+32. Each WMMA group's operands were loaded a full
// group earlier, so waits are partial and L2 latency stays covered; no
// rotation moves since each set is reloaded in the iteration it's consumed.
// Grid: 8 blocks x 8 waves cover H/16 = 64 hidden-column strips.
// h is double-buffered by the caller (other blocks may still read hprev).
// c elements are owned by exactly one lane -> safe in-place RMW.
// ---------------------------------------------------------------------------
__global__ __launch_bounds__(256) void lstm_step_fused(
    const __bf16* __restrict__ hprev,  // [B,H] bf16
    const __bf16* __restrict__ Whh,    // [4H,H] bf16 row-major
    const float* __restrict__ Gx,      // [B,4H] fp32, this timestep's slice
    float* __restrict__ c,             // [B,H] fp32 in/out
    __bf16* __restrict__ hnext,        // [B,H] bf16
    __bf16* __restrict__ yout)         // [B,H] bf16 sequence slot (or null)
{
    const int lane = threadIdx.x & 31;
    const int wave = __builtin_amdgcn_readfirstlane((int)threadIdx.x >> 5);
    const int j0   = (blockIdx.x * 8 + wave) * 16;   // hidden-column strip
    const int half = lane >> 4;
    const int l16  = lane & 15;

    const __bf16* aptr = hprev + (size_t)l16 * H_ + half * 8;
    const __bf16* wptr[4];
#pragma unroll
    for (int t = 0; t < 4; ++t)
        wptr[t] = Whh + (size_t)(t * H_ + j0 + l16) * H_ + half * 8;

    // acc[t]: gate t (i,f,g,o); register r: batch row half*8+r; lane: col j0+l16
    f32x8 acc[4];
#pragma unroll
    for (int t = 0; t < 4; ++t) {
        const int ccol = t * H_ + j0 + l16;
#pragma unroll
        for (int r = 0; r < 8; ++r)
            acc[t][r] = Gx[(size_t)(half * 8 + r) * G4_ + ccol];
    }

    // --- rolled ping-pong K pipeline ---
    bf16x16 afA, afB, wfA[4], wfB[4];
    afA = load_frag16(aptr);
#pragma unroll
    for (int t = 0; t < 4; ++t) wfA[t] = load_frag16(wptr[t]);

    int k0 = 0;
#pragma clang loop unroll(disable)
    for (; k0 + 64 < H_; k0 += 64) {
        afB = load_frag16(aptr + k0 + 32);
#pragma unroll
        for (int t = 0; t < 4; ++t) wfB[t] = load_frag16(wptr[t] + k0 + 32);
#pragma unroll
        for (int t = 0; t < 4; ++t) acc[t] = wmma_bf16(afA, wfA[t], acc[t]);

        afA = load_frag16(aptr + k0 + 64);
#pragma unroll
        for (int t = 0; t < 4; ++t) wfA[t] = load_frag16(wptr[t] + k0 + 64);
#pragma unroll
        for (int t = 0; t < 4; ++t) acc[t] = wmma_bf16(afB, wfB[t], acc[t]);
    }
    // tail: set A holds k0; one more set B at k0+32
    afB = load_frag16(aptr + k0 + 32);
#pragma unroll
    for (int t = 0; t < 4; ++t) wfB[t] = load_frag16(wptr[t] + k0 + 32);
#pragma unroll
    for (int t = 0; t < 4; ++t) acc[t] = wmma_bf16(afA, wfA[t], acc[t]);
#pragma unroll
    for (int t = 0; t < 4; ++t) acc[t] = wmma_bf16(afB, wfB[t], acc[t]);

    // in-register LSTM cell update
#pragma unroll
    for (int r = 0; r < 8; ++r) {
        const int idx = (half * 8 + r) * H_ + j0 + l16;
        const float i_ = sigmoidf_(acc[0][r]);
        const float f_ = sigmoidf_(acc[1][r]);
        const float g_ = tanhf(acc[2][r]);
        const float o_ = sigmoidf_(acc[3][r]);
        const float cn = f_ * c[idx] + i_ * g_;
        const float hn = o_ * tanhf(cn);
        c[idx] = cn;
        hnext[idx] = (__bf16)hn;
        if (yout) yout[idx] = (__bf16)hn;
    }
}

// ---------------------------------------------------------------------------
// Utilities
// ---------------------------------------------------------------------------
__global__ void cvt_f32_bf16(const float* __restrict__ src,
                             __bf16* __restrict__ dst, long n)
{
    const long i = (long)blockIdx.x * 256 + threadIdx.x;
    if (i < n) dst[i] = (__bf16)src[i];
}

__global__ void vec_add_f32(const float* __restrict__ a,
                            const float* __restrict__ b,
                            float* __restrict__ o, int n)
{
    const int i = blockIdx.x * 256 + threadIdx.x;
    if (i < n) o[i] = a[i] + b[i];
}

__global__ void fill_zero_u32(unsigned int* __restrict__ p, int n)
{
    const int i = blockIdx.x * 256 + threadIdx.x;
    if (i < n) p[i] = 0u;
}

// x[row,:] = emb_bf16[tokens[row],:]  (one 2 KB row per block)
__global__ void gather_embed(const __bf16* __restrict__ embb,
                             const int* __restrict__ tok,
                             __bf16* __restrict__ x)
{
    const int row = blockIdx.x;
    const int t = tok[row];
    const unsigned int* src = (const unsigned int*)(embb + (size_t)t * H_);
    unsigned int* dst = (unsigned int*)(x + (size_t)row * H_);
    for (int i = threadIdx.x; i < H_ / 2; i += blockDim.x) dst[i] = src[i];
}

// ---------------------------------------------------------------------------
extern "C" void kernel_launch(void* const* d_in, const int* in_sizes, int n_in,
                              void* d_out, int out_size, void* d_ws, size_t ws_size,
                              hipStream_t stream)
{
    (void)in_sizes; (void)n_in; (void)out_size; (void)ws_size;

    const int*   tokens   = (const int*)d_in[0];
    const float* emb      = (const float*)d_in[1];
    const float* dec_bias = (const float*)d_in[2];
    const float* Wih0     = (const float*)d_in[3];
    const float* Whh0     = (const float*)d_in[4];
    const float* bih0     = (const float*)d_in[5];
    const float* bhh0     = (const float*)d_in[6];
    const float* Wih1     = (const float*)d_in[7];
    const float* Whh1     = (const float*)d_in[8];
    const float* bih1     = (const float*)d_in[9];
    const float* bhh1     = (const float*)d_in[10];
    const float* latW     = (const float*)d_in[11];
    const float* latb     = (const float*)d_in[12];
    float* out = (float*)d_out;

    // ---- carve workspace (~186 MB) ----
    char* ws = (char*)d_ws;
    size_t off = 0;
    auto carve = [&](size_t bytes) -> void* {
        void* p = (void*)(ws + off);
        off += (bytes + 255) & ~(size_t)255;
        return p;
    };
    __bf16* embB  = (__bf16*)carve((size_t)V_ * H_ * 2);
    __bf16* wi0B  = (__bf16*)carve((size_t)G4_ * H_ * 2);
    __bf16* wh0B  = (__bf16*)carve((size_t)G4_ * H_ * 2);
    __bf16* wi1B  = (__bf16*)carve((size_t)G4_ * H_ * 2);
    __bf16* wh1B  = (__bf16*)carve((size_t)G4_ * H_ * 2);
    __bf16* latwB = (__bf16*)carve((size_t)H_ * H_ * 2);
    __bf16* xB    = (__bf16*)carve((size_t)SB_ * H_ * 2);
    float*  gxA   = (float*)carve((size_t)SB_ * G4_ * 4);   // x@Wih0^T + b0 (shared enc/dec L0)
    float*  gxB   = (float*)carve((size_t)SB_ * G4_ * 4);   // reused for L1 passes
    __bf16* y0B   = (__bf16*)carve((size_t)SB_ * H_ * 2);
    __bf16* d0B   = (__bf16*)carve((size_t)SB_ * H_ * 2);
    __bf16* d1B   = (__bf16*)carve((size_t)SB_ * H_ * 2);
    float*  c     = (float*)carve((size_t)B_ * H_ * 4);
    __bf16* hb0   = (__bf16*)carve((size_t)B_ * H_ * 2);    // h double buffer A
    __bf16* hb1   = (__bf16*)carve((size_t)B_ * H_ * 2);    // h double buffer B
    __bf16* he0b  = (__bf16*)carve((size_t)B_ * H_ * 2);
    __bf16* he1b  = (__bf16*)carve((size_t)B_ * H_ * 2);
    float*  ce0   = (float*)carve((size_t)B_ * H_ * 4);
    float*  ce1   = (float*)carve((size_t)B_ * H_ * 4);
    float*  dh0   = (float*)carve((size_t)B_ * H_ * 4);
    float*  dh1   = (float*)carve((size_t)B_ * H_ * 4);
    float*  bsum0 = (float*)carve((size_t)G4_ * 4);
    float*  bsum1 = (float*)carve((size_t)G4_ * 4);
    __bf16* hbuf[2] = { hb0, hb1 };

    // ---- launch helpers ----
    auto cvt = [&](const float* s, __bf16* d, long n) {
        cvt_f32_bf16<<<dim3((unsigned)((n + 255) / 256)), 256, 0, stream>>>(s, d, n);
    };
    // big GEMMs: 32x512 block tile (2 M-subtiles x 4 N-subtiles per wave)
    auto gemmBig = [&](const __bf16* A, const __bf16* W, const float* bias,
                       float* Cc, int M, int N, int K) {
        dim3 g((unsigned)((N + 511) / 512), (unsigned)(M / 32));
        wmma_gemm_bf16<2, 4><<<g, 256, 0, stream>>>(A, W, bias, Cc, M, N, K);
    };
    // small-M GEMMs (latent): 16x128 block tile, widest grid
    auto gemmSeq = [&](const __bf16* A, const __bf16* W, const float* bias,
                       float* Cc, int M, int N, int K) {
        dim3 g((unsigned)((N + 127) / 128), (unsigned)(M / 16));
        wmma_gemm_bf16<1, 1><<<g, 256, 0, stream>>>(A, W, bias, Cc, M, N, K);
    };
    // one LSTM layer pass over the full sequence (fused GEMM + cell update).
    // h starts in hbuf[0]; S_=128 even, so the final h also lands in hbuf[0].
    auto lstm_pass = [&](const __bf16* Whh, const float* Gx, __bf16* yseq) {
        for (int s = 0; s < S_; ++s) {
            lstm_step_fused<<<dim3(H_ / 128), 256, 0, stream>>>(
                hbuf[s & 1], Whh, Gx + (size_t)s * B_ * G4_, c,
                hbuf[(s + 1) & 1],
                yseq ? yseq + (size_t)s * B_ * H_ : (__bf16*)nullptr);
        }
    };
    auto zero_state = [&]() {
        fill_zero_u32<<<dim3(B_ * H_ / 256), 256, 0, stream>>>((unsigned int*)c, B_ * H_);
        fill_zero_u32<<<dim3(B_ * H_ / 512), 256, 0, stream>>>((unsigned int*)hb0, B_ * H_ / 2);
    };

    // ---- one-time bf16 conversions + bias sums + embedding gather ----
    cvt(emb,  embB,  (long)V_ * H_);
    cvt(Wih0, wi0B,  (long)G4_ * H_);
    cvt(Whh0, wh0B,  (long)G4_ * H_);
    cvt(Wih1, wi1B,  (long)G4_ * H_);
    cvt(Whh1, wh1B,  (long)G4_ * H_);
    cvt(latW, latwB, (long)H_ * H_);
    vec_add_f32<<<dim3(G4_ / 256), 256, 0, stream>>>(bih0, bhh0, bsum0, G4_);
    vec_add_f32<<<dim3(G4_ / 256), 256, 0, stream>>>(bih1, bhh1, bsum1, G4_);
    gather_embed<<<dim3(SB_), 256, 0, stream>>>(embB, tokens, xB);

    // ---- encoder layer 0 ----
    gemmBig(xB, wi0B, bsum0, gxA, SB_, G4_, H_);
    zero_state();
    lstm_pass(wh0B, gxA, y0B);
    hipMemcpyAsync(he0b, hb0, (size_t)B_ * H_ * 2, hipMemcpyDeviceToDevice, stream);
    hipMemcpyAsync(ce0,  c,   (size_t)B_ * H_ * 4, hipMemcpyDeviceToDevice, stream);

    // ---- encoder layer 1 (sequence output unused; only finals needed) ----
    gemmBig(y0B, wi1B, bsum1, gxB, SB_, G4_, H_);
    zero_state();
    lstm_pass(wh1B, gxB, nullptr);
    hipMemcpyAsync(he1b, hb0, (size_t)B_ * H_ * 2, hipMemcpyDeviceToDevice, stream);
    hipMemcpyAsync(ce1,  c,   (size_t)B_ * H_ * 4, hipMemcpyDeviceToDevice, stream);

    // ---- latent transforms: dh = he @ latW^T + latb ----
    gemmSeq(he0b, latwB, latb, dh0, B_, H_, H_);
    gemmSeq(he1b, latwB, latb, dh1, B_, H_, H_);

    // ---- decoder layer 0 (teacher-forced; reuses gxA = x@Wih0^T + b0) ----
    cvt(dh0, hb0, (long)B_ * H_);
    hipMemcpyAsync(c, ce0, (size_t)B_ * H_ * 4, hipMemcpyDeviceToDevice, stream);
    lstm_pass(wh0B, gxA, d0B);

    // ---- decoder layer 1 ----
    gemmBig(d0B, wi1B, bsum1, gxB, SB_, G4_, H_);
    cvt(dh1, hb0, (long)B_ * H_);
    hipMemcpyAsync(c, ce1, (size_t)B_ * H_ * 4, hipMemcpyDeviceToDevice, stream);
    lstm_pass(wh1B, gxB, d1B);

    // ---- tied output projection: logits = d1 @ emb^T + dec_bias ----
    gemmBig(d1B, embB, dec_bias, out, SB_, V_, H_);
}
